// Performer_32066225832680
// MI455X (gfx1250) — compile-verified, matrix operand order checked by reading, not run on hardware
//
#include <hip/hip_runtime.h>
#include <hip/hip_bf16.h>
#include <hip/hip_fp16.h>

// ---------------------------------------------------------------------------
// Performer attention on MI455X (gfx1250, wave32)
//  - all contractions on v_wmma_f32_16x16x32_f16
//  - big GEMMs: LDS-staged, double-buffered via global_load_async_to_lds_b128
// ---------------------------------------------------------------------------

typedef __attribute__((ext_vector_type(16))) _Float16 v16h;
typedef __attribute__((ext_vector_type(8)))  _Float16 v8h;
typedef __attribute__((ext_vector_type(8)))  float    v8f;

constexpr int   Bc   = 4;
constexpr int   Nn   = 2048;
constexpr int   DIM  = 768;
constexpr int   Hh   = 12;
constexpr int   Cc   = 64;
constexpr int   Mm   = 384;
constexpr int   BH   = Bc * Hh;               // 48
constexpr float SCALEF     = 0.35355339059327379f;   // 64^(-1/4)
constexpr float INV_SQRT_M = 0.051031036307982884f;  // 1/sqrt(384)
constexpr float EPSF       = 1e-8f;

// ---------------------------------------------------------------------------
// WMMA helpers
// ---------------------------------------------------------------------------

// 16x32 f16 fragment in the gfx1250 16-bit A/B layout:
// lane L: row M=(L&15), K = kb..kb+7 (v0..3) and kb+16..kb+23 (v4..7),
// kb = (L>>4)*8. Works for global or LDS pointers (b128 loads).
__device__ __forceinline__ v16h load_frag(const _Float16* __restrict__ p0, int ld) {
    const int lane = threadIdx.x & 31;
    const int row  = lane & 15;
    const int kb   = (lane >> 4) << 3;
    const _Float16* p = p0 + (size_t)row * (size_t)ld + kb;
    v8h lo = *(const v8h*)(p);
    v8h hi = *(const v8h*)(p + 16);
    v16h r;
#pragma unroll
    for (int i = 0; i < 8; ++i) { r[i] = lo[i]; r[8 + i] = hi[i]; }
    return r;
}

__device__ __forceinline__ v8f wmma_step(v16h a, v16h b, v8f c) {
    return __builtin_amdgcn_wmma_f32_16x16x32_f16(
        false, a, false, b, (short)0, c, false, false);
}

// C/D 16x16 f32 layout: lane holds col N=(lane&15); VGPR r holds row
// M = r + 8*(lane>>4).
__device__ __forceinline__ void cd_coords(int& col_in_tile, int& row0_in_tile) {
    const int lane = threadIdx.x & 31;
    col_in_tile  = lane & 15;
    row0_in_tile = (lane >> 4) << 3;
}

// Async global -> LDS copy, 16B per lane (ASYNCcnt-tracked, no VGPR data).
__device__ __forceinline__ void async_load_b128(unsigned lds_byte_off,
                                                const _Float16* gptr) {
    asm volatile("global_load_async_to_lds_b128 %0, %1, off"
                 :: "v"(lds_byte_off),
                    "v"((unsigned long long)(uintptr_t)gptr)
                 : "memory");
}
__device__ __forceinline__ void wait_async0() {
    asm volatile("s_wait_asynccnt 0x0" ::: "memory");
}

// ---------------------------------------------------------------------------
// Elementwise / conversion kernels
// ---------------------------------------------------------------------------

__global__ void cvt_f32_f16(const float* __restrict__ in, _Float16* __restrict__ out, int n) {
    int i = blockIdx.x * blockDim.x + threadIdx.x;
    if (i < n) out[i] = (_Float16)in[i];
}

// in[b][r][c] (f32) -> out[b][c][r] (f16), times scale.
__global__ void transpose_cvt_batched(const float* __restrict__ in, _Float16* __restrict__ out,
                                      int rows, int cols, float scale, int n) {
    int i = blockIdx.x * blockDim.x + threadIdx.x;
    if (i >= n) return;
    int rc  = rows * cols;
    int b   = i / rc;
    int rem = i - b * rc;
    int r   = rem / cols;
    int c   = rem - r * cols;
    out[(size_t)b * rc + (size_t)c * rows + r] = (_Float16)(in[i] * scale);
}

// sq[i] = 0.5*SCALE^2 * sum_c q[i][c]^2   (row length 64)
__global__ void rownorm_kernel(const _Float16* __restrict__ q, float* __restrict__ sq, int nrows) {
    int i = blockIdx.x * blockDim.x + threadIdx.x;
    if (i >= nrows) return;
    const _Float16* p = q + (size_t)i * Cc;
    float s = 0.f;
#pragma unroll
    for (int j = 0; j < Cc / 8; ++j) {
        v8h v = *(const v8h*)(p + j * 8);
#pragma unroll
        for (int e = 0; e < 8; ++e) { float f = (float)v[e]; s += f * f; }
    }
    sq[i] = 0.5f * SCALEF * SCALEF * s;
}

// ksum[bh][m] = sum_n kfT[bh][m][n]
__global__ void ksum_kernel(const _Float16* __restrict__ kfT, float* __restrict__ ksum, int nrows) {
    int i = blockIdx.x * blockDim.x + threadIdx.x;
    if (i >= nrows) return;
    const _Float16* p = kfT + (size_t)i * Nn;
    float s = 0.f;
    for (int j = 0; j < Nn / 8; ++j) {
        v8h v = *(const v8h*)(p + j * 8);
#pragma unroll
        for (int e = 0; e < 8; ++e) s += (float)v[e];
    }
    ksum[i] = s;
}

// D[bh][n] = eps + sum_m qf[bh][n][m] * ksum[bh][m]
__global__ void dvec_kernel(const _Float16* __restrict__ qf, const float* __restrict__ ksum,
                            float* __restrict__ Dv, int nrows) {
    int i = blockIdx.x * blockDim.x + threadIdx.x;
    if (i >= nrows) return;
    int bh = i / Nn;
    const _Float16* p = qf + (size_t)i * Mm;
    const float*    s = ksum + (size_t)bh * Mm;
    float acc = EPSF;
    for (int j = 0; j < Mm / 8; ++j) {
        v8h v = *(const v8h*)(p + j * 8);
#pragma unroll
        for (int e = 0; e < 8; ++e) acc += (float)v[e] * s[j * 8 + e];
    }
    Dv[i] = acc;
}

// ---------------------------------------------------------------------------
// LDS-staged, double-buffered, 2x2-register-blocked WMMA GEMM mainloop
//   block tile 128(M) x 64(N) x 32(K); 8 waves, wave -> 32x32 sub-tile.
//   A panels arrive via global_load_async_to_lds_b128 (ASYNCcnt),
//   fragments read back with ds_load_b128.
// ---------------------------------------------------------------------------

constexpr int GBM = 128, GBN = 64, GBK = 32;
constexpr int LDAh = 40;   // padded row length (halves): bank spread, 16B aligned

// Stage 1: qkv = xh(8192x768) @ WqkvT(2304x768 col-major), scatter epilogue
// into q[bh][n][c], k[bh][n][c], vT[bh][c][n] (f16).
__global__ void __launch_bounds__(256)
qkv_gemm_kernel(const _Float16* __restrict__ xh, const _Float16* __restrict__ WqkvT,
                _Float16* __restrict__ qh, _Float16* __restrict__ kh,
                _Float16* __restrict__ vTh) {
    __shared__ _Float16 Ash[2][GBM][LDAh];
    __shared__ _Float16 Bsh[2][GBN][LDAh];
    const int t  = threadIdx.x;
    const int w  = t >> 5;
    const int wm = w >> 1;                 // 0..3
    const int wn = w & 1;                  // 0..1
    const int rowBase = blockIdx.y * GBM;
    const int colBase = blockIdx.x * GBN;

    const unsigned ashBase = (unsigned)(uintptr_t)(&Ash[0][0][0]);
    const unsigned bshBase = (unsigned)(uintptr_t)(&Bsh[0][0][0]);
    constexpr unsigned AbufB = GBM * LDAh * 2;
    constexpr unsigned BbufB = GBN * LDAh * 2;

    auto issue = [&](int kt, int buf) {
        const int k0 = kt * GBK;
        // A panel: 128 rows x 4 chunks of 8 halves = 512 b128s; 2 per thread
#pragma unroll
        for (int rep = 0; rep < 2; ++rep) {
            const int ci  = t + rep * 256;
            const int row = ci >> 2, grp = ci & 3;
            const unsigned lds = ashBase + buf * AbufB + (unsigned)(row * LDAh + grp * 8) * 2;
            async_load_b128(lds, xh + (size_t)(rowBase + row) * DIM + k0 + grp * 8);
        }
        // B panel: 64 rows x 4 chunks = 256 b128s; 1 per thread
        {
            const int row = t >> 2, grp = t & 3;
            const unsigned lds = bshBase + buf * BbufB + (unsigned)(row * LDAh + grp * 8) * 2;
            async_load_b128(lds, WqkvT + (size_t)(colBase + row) * DIM + k0 + grp * 8);
        }
    };

    v8f acc[2][2] = {};
    constexpr int KT = DIM / GBK;          // 24
    issue(0, 0);
    for (int kt = 0; kt < KT; ++kt) {
        const int buf = kt & 1;
        wait_async0();
        __syncthreads();
        if (kt + 1 < KT) issue(kt + 1, buf ^ 1);
        v16h a0 = load_frag(&Ash[buf][wm * 32 +  0][0], LDAh);
        v16h a1 = load_frag(&Ash[buf][wm * 32 + 16][0], LDAh);
        v16h b0 = load_frag(&Bsh[buf][wn * 32 +  0][0], LDAh);
        v16h b1 = load_frag(&Bsh[buf][wn * 32 + 16][0], LDAh);
        acc[0][0] = wmma_step(a0, b0, acc[0][0]);
        acc[0][1] = wmma_step(a0, b1, acc[0][1]);
        acc[1][0] = wmma_step(a1, b0, acc[1][0]);
        acc[1][1] = wmma_step(a1, b1, acc[1][1]);
    }

    int ct, rt; cd_coords(ct, rt);
#pragma unroll
    for (int i = 0; i < 2; ++i) {
#pragma unroll
        for (int j = 0; j < 2; ++j) {
            const int col   = colBase + wn * 32 + j * 16 + ct;   // 0..2303
            const int row0  = rowBase + wm * 32 + i * 16 + rt;   // 0..8191
            const int which = col / DIM;
            const int h     = (col >> 6) % Hh;
            const int c     = col & 63;
#pragma unroll
            for (int r = 0; r < 8; ++r) {
                const int row = row0 + r;
                const int b   = row >> 11;
                const int n   = row & (Nn - 1);
                const size_t bh = (size_t)(b * Hh + h);
                const _Float16 val = (_Float16)acc[i][j][r];
                if (which == 0)      qh[(bh * Nn + n) * Cc + c]  = val;
                else if (which == 1) kh[(bh * Nn + n) * Cc + c]  = val;
                else                 vTh[(bh * Cc + c) * Nn + n] = val;
            }
        }
    }
}

// Stage 5: out = y(8192x768) @ Wproj(768x768) + bias, f32 output.
__global__ void __launch_bounds__(256)
proj_kernel(const _Float16* __restrict__ yh, const _Float16* __restrict__ WprojT,
            const float* __restrict__ bias, float* __restrict__ out) {
    __shared__ _Float16 Ash[2][GBM][LDAh];
    __shared__ _Float16 Bsh[2][GBN][LDAh];
    const int t  = threadIdx.x;
    const int w  = t >> 5;
    const int wm = w >> 1;
    const int wn = w & 1;
    const int rowBase = blockIdx.y * GBM;
    const int colBase = blockIdx.x * GBN;

    const unsigned ashBase = (unsigned)(uintptr_t)(&Ash[0][0][0]);
    const unsigned bshBase = (unsigned)(uintptr_t)(&Bsh[0][0][0]);
    constexpr unsigned AbufB = GBM * LDAh * 2;
    constexpr unsigned BbufB = GBN * LDAh * 2;

    auto issue = [&](int kt, int buf) {
        const int k0 = kt * GBK;
#pragma unroll
        for (int rep = 0; rep < 2; ++rep) {
            const int ci  = t + rep * 256;
            const int row = ci >> 2, grp = ci & 3;
            const unsigned lds = ashBase + buf * AbufB + (unsigned)(row * LDAh + grp * 8) * 2;
            async_load_b128(lds, yh + (size_t)(rowBase + row) * DIM + k0 + grp * 8);
        }
        {
            const int row = t >> 2, grp = t & 3;
            const unsigned lds = bshBase + buf * BbufB + (unsigned)(row * LDAh + grp * 8) * 2;
            async_load_b128(lds, WprojT + (size_t)(colBase + row) * DIM + k0 + grp * 8);
        }
    };

    v8f acc[2][2] = {};
    constexpr int KT = DIM / GBK;          // 24
    issue(0, 0);
    for (int kt = 0; kt < KT; ++kt) {
        const int buf = kt & 1;
        wait_async0();
        __syncthreads();
        if (kt + 1 < KT) issue(kt + 1, buf ^ 1);
        v16h a0 = load_frag(&Ash[buf][wm * 32 +  0][0], LDAh);
        v16h a1 = load_frag(&Ash[buf][wm * 32 + 16][0], LDAh);
        v16h b0 = load_frag(&Bsh[buf][wn * 32 +  0][0], LDAh);
        v16h b1 = load_frag(&Bsh[buf][wn * 32 + 16][0], LDAh);
        acc[0][0] = wmma_step(a0, b0, acc[0][0]);
        acc[0][1] = wmma_step(a0, b1, acc[0][1]);
        acc[1][0] = wmma_step(a1, b0, acc[1][0]);
        acc[1][1] = wmma_step(a1, b1, acc[1][1]);
    }

    int ct, rt; cd_coords(ct, rt);
#pragma unroll
    for (int i = 0; i < 2; ++i) {
#pragma unroll
        for (int j = 0; j < 2; ++j) {
            const int col  = colBase + wn * 32 + j * 16 + ct;
            const int row0 = rowBase + wm * 32 + i * 16 + rt;
            const float bv = bias[col];
#pragma unroll
            for (int r = 0; r < 8; ++r)
                out[(size_t)(row0 + r) * DIM + col] = acc[i][j][r] + bv;
        }
    }
}

// ---------------------------------------------------------------------------
// Batched per-head WMMA GEMMs (one 16x16 tile per wave, direct global loads)
// ---------------------------------------------------------------------------

// Stage 2: phi = exp(q @ (SCALE*w) - sq_half) / sqrt(M).  K = 64.
// transposeOut=0 -> out[bh][n][m]; 1 -> out[bh][m][n].
__global__ void phi_kernel(const _Float16* __restrict__ Qh, const _Float16* __restrict__ wT,
                           const float* __restrict__ sq, _Float16* __restrict__ out,
                           int transposeOut) {
    const int bh  = blockIdx.y;
    const int h   = bh % Hh;
    const int tid = blockIdx.x * 8 + (threadIdx.x >> 5);
    const int MT  = Mm / 16;                  // 24
    const int tm  = tid / MT;
    const int tn  = tid - tm * MT;
    const _Float16* A  = Qh + (size_t)bh * Nn * Cc + (size_t)tm * 16 * Cc;
    const _Float16* Bp = wT + (size_t)h  * Mm * Cc + (size_t)tn * 16 * Cc;
    v8f acc = {};
#pragma unroll
    for (int k = 0; k < Cc; k += 32)
        acc = wmma_step(load_frag(A + k, Cc), load_frag(Bp + k, Cc), acc);

    int ct, rt; cd_coords(ct, rt);
    const int m  = tn * 16 + ct;
    const int n0 = tm * 16 + rt;
    _Float16* obase  = out + (size_t)bh * Nn * Mm;
    const float* sqb = sq  + (size_t)bh * Nn;
#pragma unroll
    for (int r = 0; r < 8; ++r) {
        const int n = n0 + r;
        const float val = __expf(acc[r] - sqb[n]) * INV_SQRT_M;
        if (!transposeOut) obase[(size_t)n * Mm + m] = (_Float16)val;
        else               obase[(size_t)m * Nn + n] = (_Float16)val;
    }
}

// Stage 3: ktv[m][c] = sum_n kfT[m][n]*v[n][c].  K = 2048. Writes ktvT[bh][c][m].
__global__ void ktv_kernel(const _Float16* __restrict__ kfT, const _Float16* __restrict__ vTh,
                           _Float16* __restrict__ ktvT) {
    const int bh  = blockIdx.y;
    const int tid = blockIdx.x * 8 + (threadIdx.x >> 5);
    const int CT  = Cc / 16;                  // 4
    const int tm  = tid / CT;
    const int tn  = tid - tm * CT;
    const _Float16* A  = kfT + (size_t)bh * Mm * Nn + (size_t)tm * 16 * Nn;
    const _Float16* Bp = vTh + (size_t)bh * Cc * Nn + (size_t)tn * 16 * Nn;
    v8f acc = {};
#pragma unroll 4
    for (int k = 0; k < Nn; k += 32)
        acc = wmma_step(load_frag(A + k, Nn), load_frag(Bp + k, Nn), acc);

    int ct, rt; cd_coords(ct, rt);
    const int c  = tn * 16 + ct;
    const int m0 = tm * 16 + rt;
    _Float16* obase = ktvT + (size_t)bh * Cc * Mm;
#pragma unroll
    for (int r = 0; r < 8; ++r)
        obase[(size_t)c * Mm + (m0 + r)] = (_Float16)acc[r];
}

// Stage 4: out[n][c] = (qf[n] . ktv[:,c]) / D[n] -> y (B,N,768) f16.  K = 384.
__global__ void attn_out_kernel(const _Float16* __restrict__ qf, const _Float16* __restrict__ ktvT,
                                const float* __restrict__ Dv, _Float16* __restrict__ yh) {
    const int bh  = blockIdx.y;
    const int tid = blockIdx.x * 8 + (threadIdx.x >> 5);
    const int CT  = Cc / 16;                  // 4
    const int tm  = tid / CT;
    const int tn  = tid - tm * CT;
    const _Float16* A  = qf   + (size_t)bh * Nn * Mm + (size_t)tm * 16 * Mm;
    const _Float16* Bp = ktvT + (size_t)bh * Cc * Mm + (size_t)tn * 16 * Mm;
    v8f acc = {};
#pragma unroll 4
    for (int k = 0; k < Mm; k += 32)
        acc = wmma_step(load_frag(A + k, Mm), load_frag(Bp + k, Mm), acc);

    int ct, rt; cd_coords(ct, rt);
    const int c  = tn * 16 + ct;
    const int n0 = tm * 16 + rt;
    const int b  = bh / Hh;
    const int h  = bh % Hh;
    const float* Db = Dv + (size_t)bh * Nn;
#pragma unroll
    for (int r = 0; r < 8; ++r) {
        const int n = n0 + r;
        yh[((size_t)(b * Nn + n)) * DIM + h * Cc + c] = (_Float16)(acc[r] / Db[n]);
    }
}

// ---------------------------------------------------------------------------
// Host launcher
// ---------------------------------------------------------------------------

extern "C" void kernel_launch(void* const* d_in, const int* in_sizes, int n_in,
                              void* d_out, int out_size, void* d_ws, size_t ws_size,
                              hipStream_t stream) {
    (void)in_sizes; (void)n_in; (void)out_size; (void)ws_size;
    const float* x     = (const float*)d_in[0];   // (B,N,768)
    const float* w     = (const float*)d_in[1];   // (H,C,M)
    const float* Wqkv  = (const float*)d_in[2];   // (768,2304)
    const float* Wproj = (const float*)d_in[3];   // (768,768)
    const float* bproj = (const float*)d_in[4];   // (768,)
    float* out = (float*)d_out;

    char*  base = (char*)d_ws;
    size_t off  = 0;
    auto carve = [&](size_t bytes) -> char* {
        char* p = base + off;
        off += (bytes + 255) & ~(size_t)255;
        return p;
    };
    const size_t BN   = (size_t)Bc * Nn;          // 8192
    _Float16* xh      = (_Float16*)carve(BN * DIM * 2);
    _Float16* WqkvT   = (_Float16*)carve((size_t)3 * DIM * DIM * 2);
    _Float16* wT      = (_Float16*)carve((size_t)Hh * Mm * Cc * 2);
    _Float16* WprojT  = (_Float16*)carve((size_t)DIM * DIM * 2);
    _Float16* qh      = (_Float16*)carve((size_t)BH * Nn * Cc * 2);
    _Float16* kh      = (_Float16*)carve((size_t)BH * Nn * Cc * 2);
    _Float16* vTh     = (_Float16*)carve((size_t)BH * Cc * Nn * 2);
    float*    sqq     = (float*)   carve((size_t)BH * Nn * 4);
    float*    sqk     = (float*)   carve((size_t)BH * Nn * 4);
    _Float16* qf      = (_Float16*)carve((size_t)BH * Nn * Mm * 2);
    _Float16* kfT     = (_Float16*)carve((size_t)BH * Mm * Nn * 2);
    float*    ksum    = (float*)   carve((size_t)BH * Mm * 4);
    _Float16* ktvT    = (_Float16*)carve((size_t)BH * Cc * Mm * 2);
    float*    Dv      = (float*)   carve((size_t)BH * Nn * 4);
    _Float16* yh      = (_Float16*)carve(BN * DIM * 2);

    const int TB = 256;

    // 0) precision conversion / weight transposes
    {
        int n = (int)(BN * DIM);
        cvt_f32_f16<<<n / TB, TB, 0, stream>>>(x, xh, n);
    }
    {
        int n = DIM * 3 * DIM;
        transpose_cvt_batched<<<(n + TB - 1) / TB, TB, 0, stream>>>(Wqkv, WqkvT, DIM, 3 * DIM, 1.0f, n);
    }
    {
        int n = Hh * Cc * Mm;
        transpose_cvt_batched<<<(n + TB - 1) / TB, TB, 0, stream>>>(w, wT, Cc, Mm, SCALEF, n);
    }
    {
        int n = DIM * DIM;
        transpose_cvt_batched<<<(n + TB - 1) / TB, TB, 0, stream>>>(Wproj, WprojT, DIM, DIM, 1.0f, n);
    }

    // 1) QKV GEMM: block tile 128x64 -> grid (2304/64, 8192/128)
    {
        dim3 g(3 * DIM / GBN, (int)(BN / GBM));
        qkv_gemm_kernel<<<g, TB, 0, stream>>>(xh, WqkvT, qh, kh, vTh);
    }

    // 2) row norms for q, k
    {
        int nrows = BH * Nn;  // 98304
        rownorm_kernel<<<nrows / TB, TB, 0, stream>>>(qh, sqq, nrows);
        rownorm_kernel<<<nrows / TB, TB, 0, stream>>>(kh, sqk, nrows);
    }

    // 3) phi GEMMs: per bh, (2048/16)*(384/16)=3072 tiles -> 384 blocks
    {
        dim3 g(3072 / 8, BH);
        phi_kernel<<<g, TB, 0, stream>>>(qh, wT, sqq, qf, 0);   // qf: (bh, n, m)
        phi_kernel<<<g, TB, 0, stream>>>(kh, wT, sqk, kfT, 1);  // kfT: (bh, m, n)
    }

    // 4) ksum over n
    {
        int nrows = BH * Mm;  // 18432
        ksum_kernel<<<nrows / TB, TB, 0, stream>>>(kfT, ksum, nrows);
    }

    // 5) ktv GEMM: per bh, (384/16)*(64/16)=96 tiles -> 12 blocks
    {
        dim3 g(96 / 8, BH);
        ktv_kernel<<<g, TB, 0, stream>>>(kfT, vTh, ktvT);
    }

    // 6) normalizer D
    {
        int nrows = BH * Nn;
        dvec_kernel<<<nrows / TB, TB, 0, stream>>>(qf, ksum, Dv, nrows);
    }

    // 7) attention output: per bh, (2048/16)*(64/16)=512 tiles -> 64 blocks
    {
        dim3 g(512 / 8, BH);
        attn_out_kernel<<<g, TB, 0, stream>>>(qf, ktvT, Dv, yh);
    }

    // 8) final projection: block tile 128x64 -> grid (768/64, 8192/128)
    {
        dim3 g(DIM / GBN, (int)(BN / GBM));
        proj_kernel<<<g, TB, 0, stream>>>(yh, WprojT, bproj, out);
    }
}
